// GraphAttentionLayer_51238959841329
// MI455X (gfx1250) — matrix operand order repaired and checked by reading, not compiled, not run on hardware
//
#include <hip/hip_runtime.h>
#include <hip/hip_bf16.h>

#define N_NODES   8192
#define IN_FEATS  512
#define OUT_FEATS 256
#define NEG_SLOPE 0.2f

typedef __attribute__((ext_vector_type(2)))  float  v2f;
typedef __attribute__((ext_vector_type(8)))  float  v8f;
typedef __attribute__((ext_vector_type(16))) __bf16 v16bf;

// pack two f32 into one uint holding two bf16 (RNE), lo in [15:0], hi in [31:16]
__device__ __forceinline__ unsigned int bf16pk2(float lo, float hi) {
  union { float f; unsigned u; } a, b; a.f = lo; b.f = hi;
  const unsigned ua = a.u + (0x7FFFu + ((a.u >> 16) & 1u));
  const unsigned ub = b.u + (0x7FFFu + ((b.u >> 16) & 1u));
  return (ua >> 16) | (ub & 0xFFFF0000u);
}

// ---------------------------------------------------------------------------
// Kernel 1: h = x @ W in full f32 via V_WMMA_F32_16X16X4_F32.
// Each wave computes a 16x64 strip (4 N-tiles reuse one A fragment -> 4x less
// x traffic). Epilogue writes f32 h (for the score kernel) AND a transposed
// bf16 copy hT[256][8192]: each lane holds 8 consecutive rows of one column,
// so the hT write is a single packed b128 store per tile per lane.
// ---------------------------------------------------------------------------
__global__ void __launch_bounds__(256)
gat_xw_wmma_f32(const float* __restrict__ x,
                const float* __restrict__ W,
                float* __restrict__ h,
                unsigned short* __restrict__ hT) {
  const int lane = threadIdx.x & 31;
  const int wave = threadIdx.x >> 5;
  const int grp  = blockIdx.x * 8 + wave;       // 2048 wave-strips
  const int m0   = (grp >> 2) * 16;             // 512 row tiles
  const int n0   = (grp & 3) * 64;              // 4 col groups of 64
  const int half = lane >> 4;
  const int ml   = lane & 15;

  const float* xrow = x + (size_t)(m0 + ml) * IN_FEATS;

  v8f acc[4] = {};
  for (int k = 0; k < IN_FEATS; k += 4) {
    const int kb = k + 2 * half;
    v2f a;
    a.x = xrow[kb];
    a.y = xrow[kb + 1];
    const float* wr0 = W + (size_t)kb * OUT_FEATS + n0 + ml;
    const float* wr1 = wr0 + OUT_FEATS;
#pragma unroll
    for (int t = 0; t < 4; ++t) {
      v2f b;
      b.x = wr0[16 * t];
      b.y = wr1[16 * t];
      acc[t] = __builtin_amdgcn_wmma_f32_16x16x4_f32(
          false, a, false, b, (short)0, acc[t], false, false);
    }
  }

#pragma unroll
  for (int t = 0; t < 4; ++t) {
    const int col = n0 + 16 * t + ml;
    // f32 h (C/D layout: VGPR v -> row v + 8*half)
    float* hout = h + (size_t)m0 * OUT_FEATS + col;
#pragma unroll
    for (int v = 0; v < 8; ++v)
      hout[(size_t)(v + 8 * half) * OUT_FEATS] = acc[t][v];
    // transposed bf16 h: lane holds rows m0+8*half .. +7 of column `col`
    uint4 pk;
    pk.x = bf16pk2(acc[t][0], acc[t][1]);
    pk.y = bf16pk2(acc[t][2], acc[t][3]);
    pk.z = bf16pk2(acc[t][4], acc[t][5]);
    pk.w = bf16pk2(acc[t][6], acc[t][7]);
    *reinterpret_cast<uint4*>(hT + (size_t)col * N_NODES + m0 + 8 * half) = pk;
  }
}

// ---------------------------------------------------------------------------
// Kernel 2: s1 = h@a1, s2 = h@a2 — one wave32 per row, shuffle reduction.
// ---------------------------------------------------------------------------
__global__ void __launch_bounds__(256)
gat_scores(const float* __restrict__ h,
           const float* __restrict__ a,
           float* __restrict__ s1,
           float* __restrict__ s2) {
  const int lane = threadIdx.x & 31;
  const int row  = blockIdx.x * 8 + (threadIdx.x >> 5);
  const float* hr = h + (size_t)row * OUT_FEATS;
  float acc1 = 0.f, acc2 = 0.f;
  for (int j = lane; j < OUT_FEATS; j += 32) {
    const float hv = hr[j];
    acc1 += hv * a[j];
    acc2 += hv * a[OUT_FEATS + j];
  }
#pragma unroll
  for (int off = 16; off; off >>= 1) {
    acc1 += __shfl_xor(acc1, off, 32);
    acc2 += __shfl_xor(acc2, off, 32);
  }
  if (lane == 0) { s1[row] = acc1; s2[row] = acc2; }
}

// ---------------------------------------------------------------------------
// Kernel 3: per-row online-softmax stats over e = lrelu(s1_i+s2_j)*adj_ij
// (single pass over adj; adj=0 entries contribute exp(0) like the reference).
// ---------------------------------------------------------------------------
__global__ void __launch_bounds__(256)
gat_rowstats(const float* __restrict__ adj,
             const float* __restrict__ s1,
             const float* __restrict__ s2,
             float* __restrict__ rmax,
             float* __restrict__ rsum) {
  __shared__ float sm[256];
  __shared__ float ss[256];
  const int row = blockIdx.x;
  const int t   = threadIdx.x;
  const float s1v = s1[row];
  const float* ar = adj + (size_t)row * N_NODES;

  float m = -3.402823466e+38f, s = 0.f;
  for (int j = t; j < N_NODES; j += 256) {
    float e = s1v + s2[j];
    e = (e >= 0.f) ? e : NEG_SLOPE * e;
    e *= ar[j];
    if (e > m) { s = s * __expf(m - e) + 1.f; m = e; }
    else       { s += __expf(e - m); }
  }
  sm[t] = m; ss[t] = s;
  __syncthreads();
  for (int off = 128; off; off >>= 1) {
    if (t < off) {
      const float m2 = sm[t + off], sv = ss[t + off];
      const float mm = fmaxf(sm[t], m2);
      ss[t] = ss[t] * __expf(sm[t] - mm) + sv * __expf(m2 - mm);
      sm[t] = mm;
    }
    __syncthreads();
  }
  if (t == 0) { rmax[row] = sm[0]; rsum[row] = ss[0]; }
}

// ---------------------------------------------------------------------------
// Kernel 4: h' = elu( softmax(e) @ h ), fused.
// Block = 256 threads (8 waves), 16-row strip x all 256 cols (32 per wave).
// Per K=64 iteration: cooperative phase reads adj as float4 (coalesced),
// computes p = exp(e - m), packs bf16 PAIRS into LDS as uints (conversion
// done exactly once). Wave phase: A fragment = 8 ds_load_b32 (no VALU),
// B fragments = 2 global b128 loads each from the transposed bf16 hT (L2-
// resident), then 4 v_wmma_f32_16x16x32_bf16 per wave per barrier pair.
// ---------------------------------------------------------------------------
__global__ void __launch_bounds__(256)
gat_attn_h_wmma_bf16(const float* __restrict__ adj,
                     const unsigned short* __restrict__ hT,  // bf16 bits, 256 x 8192
                     const float* __restrict__ s1,
                     const float* __restrict__ s2,
                     const float* __restrict__ rmax,
                     const float* __restrict__ rsum,
                     float* __restrict__ out) {
  __shared__ unsigned int p_u32[2][16][18];   // 2 K-slabs, 16 rows, 16 used cols (pad 18)
  __shared__ float s1_sh[16];
  __shared__ float m_sh[16];

  const int i0   = blockIdx.x * 16;
  const int t    = threadIdx.x;
  const int lane = t & 31;
  const int wave = t >> 5;            // 0..7
  const int n0   = wave * 32;         // two 16-col tiles per wave

  if (t < 16) { s1_sh[t] = s1[i0 + t]; m_sh[t] = rmax[i0 + t]; }
  __syncthreads();

  const int pr   = t >> 4;            // 0..15 : p-tile row
  const int c4   = (t & 15) * 4;      // 0..60 : first of 4 p-tile cols
  const int sbuf = c4 >> 5;           // which K-slab buffer
  const int cm   = c4 & 31;
  const int half = lane >> 4;
  const int ml   = lane & 15;

  const float* adj_row = adj + (size_t)(i0 + pr) * N_NODES;
  const unsigned int* hTu = reinterpret_cast<const unsigned int*>(hT);

  v8f acc0 = {}, acc1 = {};

  for (int j0 = 0; j0 < N_NODES; j0 += 64) {
    // ---- cooperative: p = exp(e - m) for 16 x 64, packed bf16 in LDS ----
    {
      const float4 av = *reinterpret_cast<const float4*>(adj_row + j0 + c4);
      const float4 sv = *reinterpret_cast<const float4*>(s2 + j0 + c4);
      const float s1v = s1_sh[pr], mv = m_sh[pr];
      float e0 = s1v + sv.x; e0 = (e0 >= 0.f) ? e0 : NEG_SLOPE * e0; e0 *= av.x;
      float e1 = s1v + sv.y; e1 = (e1 >= 0.f) ? e1 : NEG_SLOPE * e1; e1 *= av.y;
      float e2 = s1v + sv.z; e2 = (e2 >= 0.f) ? e2 : NEG_SLOPE * e2; e2 *= av.z;
      float e3 = s1v + sv.w; e3 = (e3 >= 0.f) ? e3 : NEG_SLOPE * e3; e3 *= av.w;
      const float p0 = __expf(e0 - mv), p1 = __expf(e1 - mv);
      const float p2 = __expf(e2 - mv), p3 = __expf(e3 - mv);
      p_u32[sbuf][pr][(cm >> 1) + 0] = bf16pk2(p0, p1);
      p_u32[sbuf][pr][(cm >> 1) + 1] = bf16pk2(p2, p3);
      if (j0 + 64 < N_NODES)  // cover next slab's HBM latency
        __builtin_prefetch(adj_row + j0 + 64 + c4, 0, 1);
    }
    __syncthreads();

    // ---- wave phase: 2 K-slabs x 2 col tiles = 4 WMMAs ----
#pragma unroll
    for (int s = 0; s < 2; ++s) {
      union { unsigned int u[8]; v16bf v; } A;
#pragma unroll
      for (int v = 0; v < 8; ++v) {
        const int c = v + 8 * half;  // k0/2 of A layout
        A.u[v] = p_u32[s][ml][c];
      }
      const size_t cb = (size_t)((j0 + 32 * s) >> 1) + 8 * half;
      union { uint4 q[2]; v16bf v; } B0, B1;
      const uint4* b0p =
          reinterpret_cast<const uint4*>(hTu + (size_t)(n0 + ml) * (N_NODES / 2) + cb);
      B0.q[0] = b0p[0]; B0.q[1] = b0p[1];
      const uint4* b1p =
          reinterpret_cast<const uint4*>(hTu + (size_t)(n0 + 16 + ml) * (N_NODES / 2) + cb);
      B1.q[0] = b1p[0]; B1.q[1] = b1p[1];

      acc0 = __builtin_amdgcn_wmma_f32_16x16x32_bf16(
          false, A.v, false, B0.v, (short)0, acc0, false, false);
      acc1 = __builtin_amdgcn_wmma_f32_16x16x32_bf16(
          false, A.v, false, B1.v, (short)0, acc1, false, false);
    }
    __syncthreads();
  }

  // ---- epilogue: 1/Z scale + ELU ----
#pragma unroll
  for (int v = 0; v < 8; ++v) {
    const int row = i0 + v + 8 * half;
    const float iz = 1.0f / rsum[row];
    float v0 = acc0[v] * iz;
    float v1 = acc1[v] * iz;
    v0 = (v0 > 0.f) ? v0 : (__expf(v0) - 1.f);
    v1 = (v1 > 0.f) ? v1 : (__expf(v1) - 1.f);
    out[(size_t)row * OUT_FEATS + n0 + ml]      = v0;
    out[(size_t)row * OUT_FEATS + n0 + 16 + ml] = v1;
  }
}

// ---------------------------------------------------------------------------
extern "C" void kernel_launch(void* const* d_in, const int* in_sizes, int n_in,
                              void* d_out, int out_size, void* d_ws, size_t ws_size,
                              hipStream_t stream) {
  (void)in_sizes; (void)n_in; (void)out_size; (void)ws_size;
  const float* x   = (const float*)d_in[0];   // 8192 x 512
  const float* adj = (const float*)d_in[1];   // 8192 x 8192
  const float* W   = (const float*)d_in[2];   // 512 x 256
  const float* a   = (const float*)d_in[3];   // 512 x 1
  float* out = (float*)d_out;                 // 8192 x 256

  float* ws = (float*)d_ws;
  float*          h  = ws;                                         // 8 MB f32
  unsigned short* hT = (unsigned short*)(h + (size_t)N_NODES * OUT_FEATS);  // 4 MB bf16 (transposed)
  float*          s1 = (float*)(hT + (size_t)OUT_FEATS * N_NODES); // 8192
  float*          s2 = s1 + N_NODES;
  float*          rm = s2 + N_NODES;
  float*          rz = rm + N_NODES;

  gat_xw_wmma_f32<<<256, 256, 0, stream>>>(x, W, h, hT);
  gat_scores<<<N_NODES / 8, 256, 0, stream>>>(h, a, s1, s2);
  gat_rowstats<<<N_NODES, 256, 0, stream>>>(adj, s1, s2, rm, rz);
  gat_attn_h_wmma_bf16<<<N_NODES / 16, 256, 0, stream>>>(adj, hT, s1, s2, rm, rz, out);
}
